// Dense_11733850652928
// MI455X (gfx1250) — compile-verified, matrix operand order checked by reading, not run on hardware
//
#include <hip/hip_runtime.h>
#include <hip/hip_bf16.h>
#include <cstdint>

typedef __attribute__((ext_vector_type(16))) __bf16 v16bf;
typedef __attribute__((ext_vector_type(8)))  float  v8f;

#define B_   64
#define N_   1024
#define IN_  512
#define OUT_ 512
#define C_   8
#define KS   32            // K slice per WMMA
#define NK   (IN_ / KS)    // 16 k-iterations
#define LS   40            // padded LDS row stride in bf16 elements (80 bytes)

__device__ __forceinline__ unsigned short f2bf(float f) {
  unsigned u = __float_as_uint(f);
  u += 0x7FFFu + ((u >> 16) & 1u);   // round-to-nearest-even
  return (unsigned short)(u >> 16);
}

// Saturating fast tanh: 1 - 2/(e^{2x}+1).  exp->inf => +1, exp->0 => -1.
__device__ __forceinline__ float fast_tanh(float v) {
  float e = __expf(2.0f * v);
  return 1.0f - __fdividef(2.0f, e + 1.0f);
}

// x [B][N][IN] f32  ->  xb [N][B][IN] bf16  (atom tiles contiguous)
__global__ void cvt_x_kernel(const float* __restrict__ x,
                             unsigned short* __restrict__ xb) {
  unsigned t  = blockIdx.x * 256u + threadIdx.x;   // 8,388,608 threads
  unsigned i4 = t & 127u;                          // IN/4
  unsigned b  = (t >> 7) & 63u;
  unsigned n  = t >> 13;
  float4 v = *(const float4*)(x + ((size_t)b * N_ + n) * IN_ + i4 * 4u);
  ushort4 o;
  o.x = f2bf(v.x); o.y = f2bf(v.y); o.z = f2bf(v.z); o.w = f2bf(v.w);
  *(ushort4*)(xb + ((size_t)n * B_ + b) * IN_ + i4 * 4u) = o;
}

// weight [C][OUT][IN] f32 -> bf16, same layout (4 MiB, L2-resident)
__global__ void cvt_w_kernel(const float* __restrict__ w,
                             unsigned short* __restrict__ wb) {
  unsigned t = blockIdx.x * 256u + threadIdx.x;    // 524,288 threads
  float4 v = *(const float4*)(w + (size_t)t * 4u);
  ushort4 o;
  o.x = f2bf(v.x); o.y = f2bf(v.y); o.z = f2bf(v.z); o.w = f2bf(v.w);
  *(ushort4*)(wb + (size_t)t * 4u) = o;
}

__global__ void copy_channels_kernel(const long long* __restrict__ ch,
                                     long long* __restrict__ dst) {
  int i = blockIdx.x * 256 + threadIdx.x;
  if (i < N_) dst[i] = ch[i];
}

// grid: (OUT/128, N).  256 threads = 8 waves; wave w -> 64x16 strip.
__global__ __launch_bounds__(256, 2)
void fused_gemm_kernel(const unsigned short* __restrict__ xb,   // [N][B][IN] bf16
                       const unsigned short* __restrict__ wb,   // [C][OUT][IN] bf16
                       const float* __restrict__ x,             // residual, f32
                       const long long* __restrict__ channels,
                       const float* __restrict__ bias,          // [C][OUT] f32
                       float* __restrict__ out) {
  __shared__ __align__(16) unsigned short ldsX[2][B_ * LS];    // 10,240 B
  __shared__ __align__(16) unsigned short ldsW[2][128 * LS];   // 20,480 B

  const int atom  = blockIdx.y;
  const int obase = blockIdx.x * 128;
  const int t     = threadIdx.x;
  const int c     = (int)channels[atom];

  const unsigned short* xg = xb + (size_t)atom * B_ * IN_;
  const unsigned short* wg = wb + ((size_t)c * OUT_ + obase) * IN_;

  const int xrow = t >> 2, xkc = t & 3;

  auto issue = [&](int buf, int kk) {
    {
      unsigned lds = (unsigned)(uintptr_t)&ldsX[buf][xrow * LS + xkc * 8];
      unsigned gof = (unsigned)((xrow * IN_ + kk * KS + xkc * 8) * 2);
      asm volatile("global_load_async_to_lds_b128 %0, %1, %2"
                   :: "v"(lds), "v"(gof), "s"(xg) : "memory");
    }
#pragma unroll
    for (int cc = 0; cc < 2; ++cc) {
      int idx = cc * 256 + t;
      int row = idx >> 2, kc = idx & 3;
      unsigned lds = (unsigned)(uintptr_t)&ldsW[buf][row * LS + kc * 8];
      unsigned gof = (unsigned)((row * IN_ + kk * KS + kc * 8) * 2);
      asm volatile("global_load_async_to_lds_b128 %0, %1, %2"
                   :: "v"(lds), "v"(gof), "s"(wg) : "memory");
    }
  };

  const int wave = t >> 5, lane = t & 31;
  const int lr = lane & 15, lh = lane >> 4;

  v8f acc[4];
#pragma unroll
  for (int i = 0; i < 4; ++i)
#pragma unroll
    for (int j = 0; j < 8; ++j) acc[i][j] = 0.0f;

  union Frag { v16bf v; int4 q[2]; };

  // One k-slice of WMMA work out of buffer `buf`.
  auto compute = [&](int buf) {
    // B fragment: lane = out column within 16-wide tile; two 16B chunks per
    // the documented 16-bit layout (K = lh*8..+8 and K = 16+lh*8..+8).
    Frag bf;
    const unsigned short* pb = &ldsW[buf][(wave * 16 + lr) * LS + lh * 8];
    bf.q[0] = *(const int4*)pb;
    bf.q[1] = *(const int4*)(pb + 16);
#pragma unroll
    for (int mt = 0; mt < 4; ++mt) {
      Frag af;
      const unsigned short* pa = &ldsX[buf][(mt * 16 + lr) * LS + lh * 8];
      af.q[0] = *(const int4*)pa;
      af.q[1] = *(const int4*)(pa + 16);
      acc[mt] = __builtin_amdgcn_wmma_f32_16x16x32_bf16(
          false, af.v, false, bf.v, (short)0, acc[mt], false, false);
    }
  };

  issue(0, 0);
  // Steady state: branch-free body (always issues next slice) so the
  // accumulators stay in fixed VGPRs (no phi copies at a merge point).
  for (int kk = 0; kk < NK - 1; ++kk) {
    issue((kk + 1) & 1, kk + 1);
    asm volatile("s_wait_asynccnt 0x3" ::: "memory");  // current slice landed
    __syncthreads();
    compute(kk & 1);
    __syncthreads();
  }
  // Peeled final slice.
  asm volatile("s_wait_asynccnt 0x0" ::: "memory");
  __syncthreads();
  compute((NK - 1) & 1);

  // Epilogue: C/D layout -> col n = lane%16, row m = vgpr + 8*(lane/16)
  const int o  = obase + wave * 16 + lr;
  const float bv = bias[c * OUT_ + o];
#pragma unroll
  for (int mt = 0; mt < 4; ++mt) {
#pragma unroll
    for (int r = 0; r < 8; ++r) {
      int b = mt * 16 + lh * 8 + r;
      size_t idx = ((size_t)b * N_ + atom) * OUT_ + o;   // IN_ == OUT_
      out[idx] = fast_tanh(acc[mt][r] + bv) + x[idx];
    }
  }
}

extern "C" void kernel_launch(void* const* d_in, const int* in_sizes, int n_in,
                              void* d_out, int out_size, void* d_ws, size_t ws_size,
                              hipStream_t stream) {
  const float*     x        = (const float*)d_in[0];
  const long long* channels = (const long long*)d_in[1];
  const float*     weight   = (const float*)d_in[2];
  const float*     bias     = (const float*)d_in[3];
  float*           out      = (float*)d_out;

  unsigned short* xb = (unsigned short*)d_ws;                              // 64 MiB
  unsigned short* wb = (unsigned short*)((char*)d_ws +
                         (size_t)B_ * N_ * IN_ * sizeof(unsigned short));  // +4 MiB

  cvt_x_kernel<<<(B_ * N_ * IN_ / 4) / 256, 256, 0, stream>>>(x, xb);
  cvt_w_kernel<<<(C_ * OUT_ * IN_ / 4) / 256, 256, 0, stream>>>(weight, wb);

  dim3 grid(OUT_ / 128, N_);
  fused_gemm_kernel<<<grid, 256, 0, stream>>>(xb, wb, x, channels, bias, out);

  long long* chout = (long long*)(out + (size_t)B_ * N_ * OUT_);
  copy_channels_kernel<<<(N_ + 255) / 256, 256, 0, stream>>>(channels, chout);
}
